// BiGRU_73787538145451
// MI455X (gfx1250) — matrix operand order repaired and checked by reading, not compiled, not run on hardware
//
#include <hip/hip_runtime.h>
#include <math.h>

// ---------------------------------------------------------------------------
// BiGRU (B=64, T=512, I=H=O=1024) for gfx1250, v_wmma_f32_16x16x32_bf16.
// Persistent-scan design: the 512-step recurrence runs in ONE kernel with the
// per-workgroup Whh slice resident in LDS (192 KB of the 320 KB/WGP) and a
// grid-wide sense barrier between steps.
// ---------------------------------------------------------------------------

typedef __attribute__((ext_vector_type(16))) __bf16 v16bf;
typedef __attribute__((ext_vector_type(8)))  float  v8f;

#define B_  64
#define T_  512
#define I_  1024
#define H_  1024
#define G3_ 3072   // 3*H
#define NWG_SCAN 64          // 2 dirs * 32 column blocks
#define SMEM_SCAN 196608     // 192 frags * 1024 B (3 gates * 2 ntiles * 32 ksteps)

static __device__ inline unsigned short f2bf(float f) {
  unsigned int u = __float_as_uint(f);
  unsigned int r = (u >> 16) & 1u;               // round-to-nearest-even
  return (unsigned short)((u + 0x7fffu + r) >> 16);
}

// A fragment: 16x32 bf16, rows row0.., cols k0..k0+31, row stride lda.
// Lane L<16: row L, K=0..15; lane L>=16: row L-16, K=16..31 (contiguous).
static __device__ inline v16bf load_frag_a(const unsigned short* __restrict__ base,
                                           int lda, int row0, int k0, int lane) {
  const int r  = row0 + (lane & 15);
  const int kb = k0 + ((lane >> 4) << 4);
  const unsigned short* p = base + (size_t)r * lda + kb;
  union { uint4 q[2]; v16bf v; } u;
  u.q[0] = *(const uint4*)(p);
  u.q[1] = *(const uint4*)(p + 8);               // contiguous: elements 8..15
  return u.v;
}

// B fragment: 32x16 (KxN), B[k][n] = W[n][k], W row-major stride ldb.
// Lane L: column n0+(L&15), K = k0 + (L<16 ? 0..15 : 16..31) -- 16 CONTIGUOUS.
static __device__ inline v16bf load_frag_b(const unsigned short* __restrict__ W,
                                           int ldb, int k0, int n0, int lane) {
  const int c  = n0 + (lane & 15);
  const int kb = k0 + ((lane >> 4) << 4);
  const unsigned short* p = W + (size_t)c * ldb + kb;
  union { uint4 q[2]; v16bf v; } u;
  u.q[0] = *(const uint4*)(p);
  u.q[1] = *(const uint4*)(p + 8);               // contiguous: elements 8..15
  return u.v;
}

// Read a pre-swizzled B fragment from LDS (1024 B per fragment, 32 B per lane).
static __device__ __attribute__((always_inline)) inline v16bf
lds_frag(const unsigned char* smem, int fi, int lane) {
  const unsigned char* p = smem + fi * 1024 + lane * 32;
  union { uint4 q[2]; v16bf v; } u;
  u.q[0] = *(const uint4*)(p);
  u.q[1] = *(const uint4*)(p + 16);
  return u.v;
}

static __device__ inline v8f wmma_bf16(v16bf a, v16bf b, v8f c) {
  return __builtin_amdgcn_wmma_f32_16x16x32_bf16(
      false, a, false, b, (short)0, c, false, false);
}

static __device__ inline float sigmoidf_(float x) { return 1.0f / (1.0f + expf(-x)); }

// Sense-reversing grid barrier: cnt/gen live in d_ws, memset to 0 per call.
static __device__ inline void grid_sync(unsigned* cnt, unsigned* gen, unsigned nwg) {
  __syncthreads();
  if (threadIdx.x == 0) {
    __threadfence();
    unsigned g = __hip_atomic_load(gen, __ATOMIC_ACQUIRE, __HIP_MEMORY_SCOPE_AGENT);
    unsigned prev = __hip_atomic_fetch_add(cnt, 1u, __ATOMIC_ACQ_REL, __HIP_MEMORY_SCOPE_AGENT);
    if (prev == nwg - 1u) {
      __hip_atomic_store(cnt, 0u, __ATOMIC_RELAXED, __HIP_MEMORY_SCOPE_AGENT);
      __hip_atomic_fetch_add(gen, 1u, __ATOMIC_ACQ_REL, __HIP_MEMORY_SCOPE_AGENT);
    } else {
      while (__hip_atomic_load(gen, __ATOMIC_ACQUIRE, __HIP_MEMORY_SCOPE_AGENT) == g) {
        __builtin_amdgcn_s_sleep(1);
      }
    }
  }
  __syncthreads();
}

// ---------------------------------------------------------------------------
// fp32 -> bf16, 4 elements/thread, n divisible by 1024
// ---------------------------------------------------------------------------
__global__ __launch_bounds__(256)
void cvt_bf16_kernel(const float* __restrict__ src, unsigned short* __restrict__ dst) {
  size_t i = ((size_t)blockIdx.x * blockDim.x + threadIdx.x) * 4;
  float4 f = *(const float4*)(src + i);
  ushort4 o;
  o.x = f2bf(f.x); o.y = f2bf(f.y); o.z = f2bf(f.z); o.w = f2bf(f.w);
  *(ushort4*)(dst + i) = o;
}

// ---------------------------------------------------------------------------
// xg = x(bf16) @ Wih^T + bih -> fp32 laid out [t][b][3H].
// grid (M/256, 3H/64, 2), block 256. Wave: 32x64 strip (2 A-tiles x 4 N-tiles).
// ---------------------------------------------------------------------------
__global__ __launch_bounds__(256)
void xg_gemm_kernel(const unsigned short* __restrict__ xbf,
                    const unsigned short* __restrict__ WihF,
                    const unsigned short* __restrict__ WihB,
                    const float* __restrict__ bihF,
                    const float* __restrict__ bihB,
                    float* __restrict__ xgF,
                    float* __restrict__ xgB) {
  const int tid  = threadIdx.x;
  const int lane = tid & 31;
  const int wave = tid >> 5;
  const int dir  = blockIdx.z;
  const unsigned short* W = dir ? WihB : WihF;
  const float* bih = dir ? bihB : bihF;
  float* xg = dir ? xgB : xgF;

  const int r0 = blockIdx.x * 256 + wave * 32;   // x row r = b*T + t
  const int c0 = blockIdx.y * 64;

  const v8f zero = {0.f, 0.f, 0.f, 0.f, 0.f, 0.f, 0.f, 0.f};
  v8f acc[2][4];
#pragma unroll
  for (int m = 0; m < 2; ++m)
#pragma unroll
    for (int j = 0; j < 4; ++j) acc[m][j] = zero;

  for (int k = 0; k < I_; k += 32) {
    v16bf a0 = load_frag_a(xbf, I_, r0,      k, lane);
    v16bf a1 = load_frag_a(xbf, I_, r0 + 16, k, lane);
#pragma unroll
    for (int j = 0; j < 4; ++j) {
      v16bf b = load_frag_b(W, I_, k, c0 + 16 * j, lane);
      acc[0][j] = wmma_bf16(a0, b, acc[0][j]);
      acc[1][j] = wmma_bf16(a1, b, acc[1][j]);
    }
  }

#pragma unroll
  for (int m = 0; m < 2; ++m) {
#pragma unroll
    for (int j = 0; j < 4; ++j) {
      const int col  = c0 + 16 * j + (lane & 15);
      const float bi = bih[col];
#pragma unroll
      for (int v = 0; v < 8; ++v) {
        const int r  = r0 + m * 16 + v + ((lane >> 4) << 3);
        const int bb = r >> 9;     // r / T_
        const int tt = r & 511;    // r % T_
        xg[((size_t)tt * B_ + bb) * G3_ + col] = acc[m][j][v] + bi;
      }
    }
  }
}

// ---------------------------------------------------------------------------
// Persistent GRU scan: grid = 64 WGs (wg>>5 = dir, wg&31 = 32-col block),
// block 256 (8 waves: mt = w&3 row-tile, nt = w>>2 col-tile).
// Stage Whh slice into LDS fragments once, then loop t = 0..511 with a grid
// barrier between steps. Each wave owns the r/z/n accumulators of one 16x16
// output tile -> gate math is purely per-lane.
// ---------------------------------------------------------------------------
__global__ __launch_bounds__(256)
void gru_scan_kernel(const unsigned short* __restrict__ WhhF,
                     const unsigned short* __restrict__ WhhB,
                     const float* __restrict__ bhhF,
                     const float* __restrict__ bhhB,
                     const float* __restrict__ xgF,
                     const float* __restrict__ xgB,
                     unsigned short* __restrict__ hbf,   // [dir][2][B*H] bf16
                     float* __restrict__ hf,             // [dir][2][B*H] f32
                     unsigned* __restrict__ bar) {       // {cnt, gen}
  extern __shared__ unsigned char smem[];

  const int tid  = threadIdx.x;
  const int lane = tid & 31;
  const int wave = tid >> 5;
  const int wg   = blockIdx.x;
  const int dir  = wg >> 5;
  const int j0wg = (wg & 31) * 32;
  const int mt   = wave & 3;
  const int nt   = wave >> 2;

  const unsigned short* Whh = dir ? WhhB : WhhF;
  const float* bhh = dir ? bhhB : bhhF;
  const float* xg  = dir ? xgB : xgF;

  // ---- Stage this WG's Whh slice into LDS in fragment layout (192 KB) ----
  // frag fi = (g*2 + nt)*32 + ks; 1024 B each; lane stores its 32 bytes.
  for (int fi = wave; fi < 192; fi += 8) {
    const int g  = fi >> 6;
    const int fn = (fi >> 5) & 1;
    const int ks = fi & 31;
    const int col = g * H_ + j0wg + fn * 16 + (lane & 15);
    const int kb  = ks * 32 + ((lane >> 4) << 4);
    const unsigned short* p = Whh + (size_t)col * H_ + kb;
    uint4 q0 = *(const uint4*)(p);
    uint4 q1 = *(const uint4*)(p + 8);
    unsigned char* d = smem + fi * 1024 + lane * 32;
    *(uint4*)(d)      = q0;
    *(uint4*)(d + 16) = q1;
  }
  __syncthreads();

  const int j0  = j0wg + nt * 16;
  const int col = j0 + (lane & 15);
  const float br = bhh[col], bz = bhh[H_ + col], bn = bhh[2 * H_ + col];
  const size_t HB = (size_t)B_ * H_;
  const v8f zero = {0.f, 0.f, 0.f, 0.f, 0.f, 0.f, 0.f, 0.f};

  for (int t = 0; t < T_; ++t) {
    const int cur = t & 1, nxt = cur ^ 1;
    const unsigned short* hbf_cur = hbf + (size_t)(dir * 2 + cur) * HB;
    unsigned short*       hbf_nxt = hbf + (size_t)(dir * 2 + nxt) * HB;
    const float*          hf_cur  = hf  + (size_t)(dir * 2 + cur) * HB;
    float*                hf_nxt  = hf  + (size_t)(dir * 2 + nxt) * HB;

    v8f accR = zero, accZ = zero, accN = zero;
    for (int ks = 0; ks < 32; ++ks) {
      v16bf a  = load_frag_a(hbf_cur, H_, mt * 16, ks * 32, lane);
      v16bf bR = lds_frag(smem, (0 * 2 + nt) * 32 + ks, lane);
      v16bf bZ = lds_frag(smem, (1 * 2 + nt) * 32 + ks, lane);
      v16bf bN = lds_frag(smem, (2 * 2 + nt) * 32 + ks, lane);
      accR = wmma_bf16(a, bR, accR);
      accZ = wmma_bf16(a, bZ, accZ);
      accN = wmma_bf16(a, bN, accN);
    }

    const int teff = dir ? (T_ - 1 - t) : t;
#pragma unroll
    for (int v = 0; v < 8; ++v) {
      const int b = mt * 16 + v + ((lane >> 4) << 3);
      const float* xrow = xg + ((size_t)teff * B_ + b) * G3_;
      const float xr = xrow[col];
      const float xz = xrow[H_ + col];
      const float xn = xrow[2 * H_ + col];
      const float r  = sigmoidf_(xr + accR[v] + br);
      const float z  = sigmoidf_(xz + accZ[v] + bz);
      const float n  = tanhf(xn + r * (accN[v] + bn));
      const float hp = hf_cur[(size_t)b * H_ + col];
      const float hnew = (1.0f - z) * n + z * hp;
      hf_nxt[(size_t)b * H_ + col]  = hnew;
      hbf_nxt[(size_t)b * H_ + col] = f2bf(hnew);
    }

    grid_sync(bar, bar + 1, NWG_SCAN);
  }
}

// ---------------------------------------------------------------------------
// out[b][o] = b_lin[o] + h_f[b]·W_lin[o][0:H] + h_b[b]·W_lin[o][H:2H]
// ---------------------------------------------------------------------------
__global__ __launch_bounds__(256)
void linear_kernel(const float* __restrict__ hF, const float* __restrict__ hB,
                   const float* __restrict__ Wlin, const float* __restrict__ blin,
                   float* __restrict__ out) {
  const int idx = blockIdx.x * blockDim.x + threadIdx.x;
  const int b = idx >> 10;
  const int o = idx & 1023;
  const float* w = Wlin + (size_t)o * (2 * H_);
  float s = blin[o];
  const float* hfp = hF + (size_t)b * H_;
  const float* hbp = hB + (size_t)b * H_;
  for (int j = 0; j < H_; ++j) s += hfp[j] * w[j];
  for (int j = 0; j < H_; ++j) s += hbp[j] * w[H_ + j];
  out[idx] = s;
}

// ---------------------------------------------------------------------------
extern "C" void kernel_launch(void* const* d_in, const int* in_sizes, int n_in,
                              void* d_out, int out_size, void* d_ws, size_t ws_size,
                              hipStream_t stream) {
  const float* x    = (const float*)d_in[0];
  const float* WihF = (const float*)d_in[1];
  const float* WhhF = (const float*)d_in[2];
  const float* bihF = (const float*)d_in[3];
  const float* bhhF = (const float*)d_in[4];
  const float* WihB = (const float*)d_in[5];
  const float* WhhB = (const float*)d_in[6];
  const float* bihB = (const float*)d_in[7];
  const float* bhhB = (const float*)d_in[8];
  const float* Wlin = (const float*)d_in[9];
  const float* blin = (const float*)d_in[10];
  float* out = (float*)d_out;

  char* ws = (char*)d_ws;
  size_t off = 0;
  auto alloc = [&](size_t bytes) -> char* {
    char* p = ws + off;
    off = (off + bytes + 255) & ~(size_t)255;
    return p;
  };
  unsigned short* xbf    = (unsigned short*)alloc((size_t)B_ * T_ * I_ * 2);
  unsigned short* WihFbf = (unsigned short*)alloc((size_t)G3_ * I_ * 2);
  unsigned short* WihBbf = (unsigned short*)alloc((size_t)G3_ * I_ * 2);
  unsigned short* WhhFbf = (unsigned short*)alloc((size_t)G3_ * H_ * 2);
  unsigned short* WhhBbf = (unsigned short*)alloc((size_t)G3_ * H_ * 2);
  float* xgF = (float*)alloc((size_t)T_ * B_ * G3_ * 4);
  float* xgB = (float*)alloc((size_t)T_ * B_ * G3_ * 4);
  float* hf           = (float*)alloc((size_t)2 * 2 * B_ * H_ * 4);
  unsigned short* hbf = (unsigned short*)alloc((size_t)2 * 2 * B_ * H_ * 2);
  unsigned* bar       = (unsigned*)alloc(256);

  // 1) fp32 -> bf16
  cvt_bf16_kernel<<<(B_ * T_ * I_) / 1024, 256, 0, stream>>>(x, xbf);
  cvt_bf16_kernel<<<(G3_ * I_) / 1024, 256, 0, stream>>>(WihF, WihFbf);
  cvt_bf16_kernel<<<(G3_ * I_) / 1024, 256, 0, stream>>>(WihB, WihBbf);
  cvt_bf16_kernel<<<(G3_ * H_) / 1024, 256, 0, stream>>>(WhhF, WhhFbf);
  cvt_bf16_kernel<<<(G3_ * H_) / 1024, 256, 0, stream>>>(WhhB, WhhBbf);

  // 2) Input-projection GEMM, both directions
  xg_gemm_kernel<<<dim3((B_ * T_) / 256, G3_ / 64, 2), 256, 0, stream>>>(
      xbf, WihFbf, WihBbf, bihF, bihB, xgF, xgB);

  // 3) Zero h0 (ping-pong buffers) and barrier state (capture-safe)
  hipMemsetAsync(hf,  0, (size_t)2 * 2 * B_ * H_ * 4, stream);
  hipMemsetAsync(hbf, 0, (size_t)2 * 2 * B_ * H_ * 2, stream);
  hipMemsetAsync(bar, 0, 256, stream);

  // 4) Persistent bidirectional scan (one launch, 512 steps, grid barrier)
  gru_scan_kernel<<<NWG_SCAN, 256, SMEM_SCAN, stream>>>(
      WhhFbf, WhhBbf, bhhF, bhhB, xgF, xgB, hbf, hf, bar);

  // 5) Output linear; after 512 steps final h is in ping-pong buffer 0.
  const size_t HB = (size_t)B_ * H_;
  const float* hF_last = hf + (size_t)(0 * 2 + (T_ & 1)) * HB;
  const float* hB_last = hf + (size_t)(1 * 2 + (T_ & 1)) * HB;
  linear_kernel<<<(B_ * 1024) / 256, 256, 0, stream>>>(hF_last, hB_last, Wlin, blin, out);
}
